// MotionFormerBlock_5291399708691
// MI455X (gfx1250) — matrix operand order repaired and checked by reading, not compiled, not run on hardware
//
#include <hip/hip_runtime.h>

// ---------------- types ----------------
typedef __attribute__((ext_vector_type(16))) __bf16 v16bf;
typedef __attribute__((ext_vector_type(8)))  __bf16 v8bf;
typedef __attribute__((ext_vector_type(8)))  float  v8f;

#define BB   8
#define NN   1024
#define CC   256
#define MD   128
#define HH   8
#define HD   32
#define MHD  16
#define HIDD 1024
#define TOK  (BB * NN)          // 8192 tokens
#define SCALE 0.17677669529663687f  // 1/sqrt(32)

// ---------------- fragment loads ----------------
static __device__ inline v16bf ld16(const __bf16* p) {
  v8bf lo = *reinterpret_cast<const v8bf*>(p);
  v8bf hi = *reinterpret_cast<const v8bf*>(p + 8);
  v16bf r;
#pragma unroll
  for (int i = 0; i < 8; ++i) { r[i] = lo[i]; r[i + 8] = hi[i]; }
  return r;
}
static __device__ inline v16bf ld8x2(const __bf16* a, const __bf16* b) {
  v8bf lo = *reinterpret_cast<const v8bf*>(a);
  v8bf hi = *reinterpret_cast<const v8bf*>(b);
  v16bf r;
#pragma unroll
  for (int i = 0; i < 8; ++i) { r[i] = lo[i]; r[i + 8] = hi[i]; }
  return r;
}

// ---------------- f32 -> bf16 copy ----------------
__global__ void f2b_kernel(const float* __restrict__ in, __bf16* __restrict__ out, int n) {
  int i = blockIdx.x * 256 + threadIdx.x;
  if (i < n) out[i] = (__bf16)in[i];
}

// ---------------- BatchNorm stats ----------------
// bn1: x is [B, C, N]; stats per channel c over (B,N)
__global__ void bn1_stats_kernel(const float* __restrict__ x, float* __restrict__ mean,
                                 float* __restrict__ rstd) {
  __shared__ float ss[256], sq[256];
  int c = blockIdx.x, tid = threadIdx.x;
  float s = 0.f, q = 0.f;
  for (int i = tid; i < TOK; i += 256) {
    int b = i >> 10, n = i & 1023;
    float v = x[((size_t)(b * CC + c)) * NN + n];
    s += v; q += v * v;
  }
  ss[tid] = s; sq[tid] = q; __syncthreads();
  for (int o = 128; o > 0; o >>= 1) {
    if (tid < o) { ss[tid] += ss[tid + o]; sq[tid] += sq[tid + o]; }
    __syncthreads();
  }
  if (tid == 0) {
    float m = ss[0] * (1.f / TOK);
    float v = sq[0] * (1.f / TOK) - m * m;
    mean[c] = m; rstd[c] = rsqrtf(v + 1e-5f);
  }
}
// bn2: xn2 is [T, C] token-major; stats per channel c over T
__global__ void bn2_stats_kernel(const float* __restrict__ xn2, float* __restrict__ mean,
                                 float* __restrict__ rstd) {
  __shared__ float ss[256], sq[256];
  int c = blockIdx.x, tid = threadIdx.x;
  float s = 0.f, q = 0.f;
  for (int t = tid; t < TOK; t += 256) {
    float v = xn2[(size_t)t * CC + c];
    s += v; q += v * v;
  }
  ss[tid] = s; sq[tid] = q; __syncthreads();
  for (int o = 128; o > 0; o >>= 1) {
    if (tid < o) { ss[tid] += ss[tid + o]; sq[tid] += sq[tid + o]; }
    __syncthreads();
  }
  if (tid == 0) {
    float m = ss[0] * (1.f / TOK);
    float v = sq[0] * (1.f / TOK) - m * m;
    mean[c] = m; rstd[c] = rsqrtf(v + 1e-5f);
  }
}

// bn1 apply: xn [T,C] f32 + bf16 (tokens)
__global__ void bn1_apply_kernel(const float* __restrict__ x, const float* __restrict__ mean,
                                 const float* __restrict__ rstd, const float* __restrict__ g,
                                 const float* __restrict__ bb, float* __restrict__ xnf,
                                 __bf16* __restrict__ xnb) {
  int i = blockIdx.x * 256 + threadIdx.x;       // i = t*256 + c
  int t = i >> 8, c = i & 255;
  int b = t >> 10, n = t & 1023;
  float v = (x[((size_t)(b * CC + c)) * NN + n] - mean[c]) * rstd[c] * g[c] + bb[c];
  xnf[i] = v; xnb[i] = (__bf16)v;
}
// bn2 apply: xb bf16 only
__global__ void bn2_apply_kernel(const float* __restrict__ xn2, const float* __restrict__ mean,
                                 const float* __restrict__ rstd, const float* __restrict__ g,
                                 const float* __restrict__ bb, __bf16* __restrict__ xb) {
  int i = blockIdx.x * 256 + threadIdx.x;
  int c = i & 255;
  xb[i] = (__bf16)((xn2[i] - mean[c]) * rstd[c] * g[c] + bb[c]);
}

// ---------------- coordinate embedding ce = cor @ cor_w.T (K=3) ----------------
__global__ void ce_kernel(const float* __restrict__ cor, const float* __restrict__ cor_w,
                          float* __restrict__ cef) {
  int i = blockIdx.x * 256 + threadIdx.x;       // i = t*128 + m
  int t = i >> 7, m = i & 127;
  const float* cr = cor + (size_t)t * 3;
  const float* w  = cor_w + (size_t)m * 3;
  cef[i] = cr[0] * w[0] + cr[1] * w[1] + cr[2] * w[2];
}

// ---------------- transposes for attention B-operands ----------------
// Vt[b][h][d][n] from kvb[t][256 + h*32 + d]
__global__ void vt_kernel(const __bf16* __restrict__ kvb, __bf16* __restrict__ vt) {
  int i = blockIdx.x * 256 + threadIdx.x;       // b,h,d,n
  int n = i & 1023; int r = i >> 10;
  int d = r & 31; r >>= 5;
  int h = r & 7;  int b = r >> 3;
  vt[i] = kvb[((size_t)(b * NN + n)) * (2 * CC) + CC + h * HD + d];
}
// CEHt[b][h][d][n] (bf16) from cef[t][h*16+d]
__global__ void ceht_kernel(const float* __restrict__ cef, __bf16* __restrict__ ct) {
  int i = blockIdx.x * 256 + threadIdx.x;
  int n = i & 1023; int r = i >> 10;
  int d = r & 15; r >>= 4;
  int h = r & 7;  int b = r >> 3;
  ct[i] = (__bf16)cef[((size_t)(b * NN + n)) * MD + h * MHD + d];
}

// ---------------- fused attention (two-pass softmax, WMMA) ----------------
// grid: (B*H, N/16), block: 32 (one wave). 64KB dynamic LDS for 16x1024 scores.
// Computes S^T (A = K-tile, B = Q-tile) so the f32 D layout maps directly onto
// the bf16 A-operand slot layout of the following P@V WMMA (no lane transpose).
__global__ __launch_bounds__(32)
void attn_kernel(const __bf16* __restrict__ qb, const __bf16* __restrict__ kvb,
                 const __bf16* __restrict__ vt, const __bf16* __restrict__ ceht,
                 const float* __restrict__ cef, __bf16* __restrict__ xab,
                 __bf16* __restrict__ motb) {
  extern __shared__ float S[];                  // S[key*16 + q], 1024x16 f32
  int lane = threadIdx.x;
  int half = lane >> 4;                         // 0 or 1
  int qq   = lane & 15;
  int bh = blockIdx.x; int b = bh >> 3; int h = bh & 7;
  int q0 = blockIdx.y * 16;

  // Q as B-operand of S^T = K @ Q^T : element j = Q[q0+qq][h*32 + 16*half + j]
  const __bf16* qrow = qb + ((size_t)(b * NN + q0 + qq)) * CC + h * HD + 16 * half;
  v16bf qf = ld16(qrow);

  // Pass 1: scores for all 1024 keys, 16 keys per WMMA
  for (int c = 0; c < NN / 16; ++c) {
    int k0 = c * 16;
    const __bf16* krow = kvb + ((size_t)(b * NN + k0 + qq)) * (2 * CC) + h * HD;
    v16bf kf = ld16(krow + 16 * half);
    v8f z = {};
    v8f st = __builtin_amdgcn_wmma_f32_16x16x32_bf16(false, kf, false, qf, (short)0, z,
                                                     false, false);
#pragma unroll
    for (int r = 0; r < 8; ++r)
      S[(k0 + r + 8 * half) * 16 + qq] = st[r] * SCALE;
  }

  // Pass 2: softmax stats for query qq; lane half scans keys [half*512, half*512+512)
  int kb = half * 512;
  float m = -3.0e38f;
  for (int k = 0; k < 512; ++k) m = fmaxf(m, S[(kb + k) * 16 + qq]);
  m = fmaxf(m, __shfl_xor(m, 16, 32));
  float s = 0.f;
  for (int k = 0; k < 512; ++k) s += __expf(S[(kb + k) * 16 + qq] - m);
  s += __shfl_xor(s, 16, 32);
  float rs = 1.f / s;

  // Pass 3: O = P @ [V | ceh], 32 keys per iteration (1 P-frag, 3 WMMAs)
  v8f o0 = {}, o1 = {}, oc = {};
  const __bf16* vt0 = vt   + (((size_t)(b * HH + h) * HD) + qq)      * NN;
  const __bf16* vt1 = vt   + (((size_t)(b * HH + h) * HD) + 16 + qq) * NN;
  const __bf16* ct0 = ceht + (((size_t)(b * HH + h) * MHD) + qq)     * NN;
  for (int cc = 0; cc < NN / 32; ++cc) {
    int kbase = cc * 32;
    v16bf p;
#pragma unroll
    for (int j = 0; j < 16; ++j) {
      int key = kbase + 16 * half + j;
      p[j] = (__bf16)(__expf(S[key * 16 + qq] - m) * rs);
    }
    int ko = kbase + 16 * half;
    o0 = __builtin_amdgcn_wmma_f32_16x16x32_bf16(false, p, false, ld16(vt0 + ko), (short)0,
                                                 o0, false, false);
    o1 = __builtin_amdgcn_wmma_f32_16x16x32_bf16(false, p, false, ld16(vt1 + ko), (short)0,
                                                 o1, false, false);
    oc = __builtin_amdgcn_wmma_f32_16x16x32_bf16(false, p, false, ld16(ct0 + ko), (short)0,
                                                 oc, false, false);
  }

  // Epilogue: D layout lane = out-dim column, rows = q0 + r + 8*half
#pragma unroll
  for (int r = 0; r < 8; ++r) {
    int t = b * NN + q0 + r + 8 * half;
    xab[(size_t)t * CC + h * HD + qq]      = (__bf16)o0[r];
    xab[(size_t)t * CC + h * HD + 16 + qq] = (__bf16)o1[r];
    size_t mi = (size_t)t * MD + h * MHD + qq;
    motb[mi] = (__bf16)(oc[r] - cef[mi]);
  }
}

// ---------------- generic WMMA GEMM: C[M,N] = A[M,K] @ W[N,K]^T (+epilogue) ----------------
// 2x2 register blocking: each wave owns a 32x32 output tile (4 accumulators);
// per 32-wide k-step: 2 A-fragments + 2 W-fragments feed 4 WMMAs.
enum { MODE_OUT_BF16 = 0, MODE_RES_F32 = 1, MODE_OUT_F32 = 2, MODE_DW_PRELU = 3,
       MODE_FINAL_X = 4 };

struct GemmP {
  const __bf16* A; const __bf16* W; const float* bias;
  int M, N, K, swapRows, mode;
  float* outF; __bf16* outB;
  const float* res;                              // MODE_RES_F32
  const float* dww; const float* dwb; const float* prelu;  // MODE_DW_PRELU
  const float* xin;                              // MODE_FINAL_X
};

__global__ __launch_bounds__(32) void gemm_wmma_kernel(GemmP p) {
  int lane = threadIdx.x;
  int half = lane >> 4, ln = lane & 15;
  int m0 = blockIdx.x * 32, n0 = blockIdx.y * 32;

  // A-operand rows for the two m-tiles
  int rowA0 = m0 + ln, rowA1 = m0 + 16 + ln;
  if (p.swapRows) {
    int b0 = rowA0 >> 10, nn0 = rowA0 & 1023; rowA0 = (((b0 + 4) & 7) << 10) | nn0;
    int b1 = rowA1 >> 10, nn1 = rowA1 & 1023; rowA1 = (((b1 + 4) & 7) << 10) | nn1;
  }
  const __bf16* Arow0 = p.A + (size_t)rowA0 * p.K;
  const __bf16* Arow1 = p.A + (size_t)rowA1 * p.K;
  const __bf16* Wrow0 = p.W + (size_t)(n0 + ln) * p.K + 16 * half;
  const __bf16* Wrow1 = p.W + (size_t)(n0 + 16 + ln) * p.K + 16 * half;

  v8f a00 = {}, a01 = {}, a10 = {}, a11 = {};
  for (int k = 0; k < p.K; k += 32) {
    v16bf af0 = ld16(Arow0 + k + 16 * half);
    v16bf af1 = ld16(Arow1 + k + 16 * half);
    v16bf wf0 = ld16(Wrow0 + k);
    v16bf wf1 = ld16(Wrow1 + k);
    a00 = __builtin_amdgcn_wmma_f32_16x16x32_bf16(false, af0, false, wf0, (short)0, a00,
                                                  false, false);
    a01 = __builtin_amdgcn_wmma_f32_16x16x32_bf16(false, af0, false, wf1, (short)0, a01,
                                                  false, false);
    a10 = __builtin_amdgcn_wmma_f32_16x16x32_bf16(false, af1, false, wf0, (short)0, a10,
                                                  false, false);
    a11 = __builtin_amdgcn_wmma_f32_16x16x32_bf16(false, af1, false, wf1, (short)0, a11,
                                                  false, false);
  }

#pragma unroll
  for (int ni = 0; ni < 2; ++ni) {
    int col = n0 + ni * 16 + ln;
    float bs = p.bias ? p.bias[col] : 0.f;
#pragma unroll
    for (int mi = 0; mi < 2; ++mi) {
      v8f acc = (mi == 0) ? (ni == 0 ? a00 : a01) : (ni == 0 ? a10 : a11);
#pragma unroll
      for (int r = 0; r < 8; ++r) {
        int row = m0 + mi * 16 + r + 8 * half;
        float v = acc[r] + bs;
        if (p.mode == MODE_OUT_BF16) {
          p.outB[(size_t)row * p.N + col] = (__bf16)v;
        } else if (p.mode == MODE_RES_F32) {
          size_t idx = (size_t)row * p.N + col;
          p.outF[idx] = v + p.res[idx];
        } else if (p.mode == MODE_OUT_F32) {
          p.outF[(size_t)row * p.N + col] = v;
        } else if (p.mode == MODE_DW_PRELU) {
          v = v * p.dww[col] + p.dwb[col];
          float a0 = p.prelu[0];
          v = (v >= 0.f) ? v : a0 * v;
          p.outB[(size_t)row * p.N + col] = (__bf16)v;
        } else { // MODE_FINAL_X: x_out[b, col, n] = x[b, col, n] + v
          int b = row >> 10, n = row & 1023;
          size_t idx = ((size_t)(b * CC + col)) * NN + n;
          p.outF[idx] = p.xin[idx] + v;
        }
      }
    }
  }
}

// ---------------- host launcher ----------------
extern "C" void kernel_launch(void* const* d_in, const int* in_sizes, int n_in,
                              void* d_out, int out_size, void* d_ws, size_t ws_size,
                              hipStream_t stream) {
  const float* x      = (const float*)d_in[0];
  const float* cor    = (const float*)d_in[1];
  const float* q_w    = (const float*)d_in[2];
  const float* kv_w   = (const float*)d_in[3];
  const float* cor_w  = (const float*)d_in[4];
  const float* proj_w = (const float*)d_in[5];
  const float* proj_b = (const float*)d_in[6];
  const float* mproj_w= (const float*)d_in[7];
  const float* mproj_b= (const float*)d_in[8];
  const float* g1     = (const float*)d_in[9];
  const float* b1     = (const float*)d_in[10];
  const float* g2     = (const float*)d_in[11];
  const float* b2     = (const float*)d_in[12];
  const float* fc1_w  = (const float*)d_in[13];
  const float* fc1_b  = (const float*)d_in[14];
  const float* dw_w   = (const float*)d_in[15];
  const float* dw_b   = (const float*)d_in[16];
  const float* prelu  = (const float*)d_in[17];
  const float* fc2_w  = (const float*)d_in[18];
  const float* fc2_b  = (const float*)d_in[19];

  float* out_x   = (float*)d_out;                        // [8,256,1024]
  float* out_mot = (float*)d_out + (size_t)BB * CC * NN; // [8,1024,128]

  // workspace carve-out (256B aligned)
  char* w = (char*)d_ws; size_t off = 0;
  auto alloc = [&](size_t bytes) -> void* {
    void* p = w + off; off = (off + bytes + 255) & ~(size_t)255; return p;
  };
  float* st1m = (float*)alloc(CC * 4);
  float* st1r = (float*)alloc(CC * 4);
  float* st2m = (float*)alloc(CC * 4);
  float* st2r = (float*)alloc(CC * 4);
  __bf16* wq  = (__bf16*)alloc((size_t)CC * CC * 2);
  __bf16* wkv = (__bf16*)alloc((size_t)2 * CC * CC * 2);
  __bf16* wpj = (__bf16*)alloc((size_t)CC * CC * 2);
  __bf16* wmp = (__bf16*)alloc((size_t)MD * MD * 2);
  __bf16* wf1 = (__bf16*)alloc((size_t)HIDD * CC * 2);
  __bf16* wf2 = (__bf16*)alloc((size_t)CC * HIDD * 2);
  float*  xnf = (float*)alloc((size_t)TOK * CC * 4);
  __bf16* xnb = (__bf16*)alloc((size_t)TOK * CC * 2);
  __bf16* qb  = (__bf16*)alloc((size_t)TOK * CC * 2);
  __bf16* kvb = (__bf16*)alloc((size_t)TOK * 2 * CC * 2);
  __bf16* vt  = (__bf16*)alloc((size_t)BB * HH * HD * NN * 2);
  float*  cef = (float*)alloc((size_t)TOK * MD * 4);
  __bf16* cht = (__bf16*)alloc((size_t)BB * HH * MHD * NN * 2);
  __bf16* xab = (__bf16*)alloc((size_t)TOK * CC * 2);
  __bf16* mob = (__bf16*)alloc((size_t)TOK * MD * 2);
  float*  xn2 = (float*)alloc((size_t)TOK * CC * 4);
  __bf16* xbb = (__bf16*)alloc((size_t)TOK * CC * 2);
  __bf16* hb  = (__bf16*)alloc((size_t)TOK * HIDD * 2);
  (void)ws_size; (void)n_in; (void)in_sizes; (void)out_size;

  auto cvt = [&](const float* src, __bf16* dst, int n) {
    f2b_kernel<<<(n + 255) / 256, 256, 0, stream>>>(src, dst, n);
  };
  cvt(q_w,    wq,  CC * CC);
  cvt(kv_w,   wkv, 2 * CC * CC);
  cvt(proj_w, wpj, CC * CC);
  cvt(mproj_w,wmp, MD * MD);
  cvt(fc1_w,  wf1, HIDD * CC);
  cvt(fc2_w,  wf2, CC * HIDD);

  // BN1 + tokens
  bn1_stats_kernel<<<CC, 256, 0, stream>>>(x, st1m, st1r);
  bn1_apply_kernel<<<TOK * CC / 256, 256, 0, stream>>>(x, st1m, st1r, g1, b1, xnf, xnb);

  GemmP gp{};
  // q = xn @ q_w.T   -> bf16 [T,256]
  gp = GemmP{xnb, wq, nullptr, TOK, CC, CC, 0, MODE_OUT_BF16,
             nullptr, qb, nullptr, nullptr, nullptr, nullptr, nullptr};
  gemm_wmma_kernel<<<dim3(TOK / 32, CC / 32), 32, 0, stream>>>(gp);
  // kv = xr @ kv_w.T -> bf16 [T,512] (batch-half swap on A rows)
  gp = GemmP{xnb, wkv, nullptr, TOK, 2 * CC, CC, 1, MODE_OUT_BF16,
             nullptr, kvb, nullptr, nullptr, nullptr, nullptr, nullptr};
  gemm_wmma_kernel<<<dim3(TOK / 32, 2 * CC / 32), 32, 0, stream>>>(gp);

  // ce + transposed attention operands
  ce_kernel<<<TOK * MD / 256, 256, 0, stream>>>(cor, cor_w, cef);
  vt_kernel<<<BB * HH * HD * NN / 256, 256, 0, stream>>>(kvb, vt);
  ceht_kernel<<<BB * HH * MHD * NN / 256, 256, 0, stream>>>(cef, cht);

  // fused attention (value + motion paths)
  attn_kernel<<<dim3(BB * HH, NN / 16), 32, 16 * NN * sizeof(float), stream>>>(
      qb, kvb, vt, cht, cef, xab, mob);

  // proj + residual -> xn2 f32
  gp = GemmP{xab, wpj, proj_b, TOK, CC, CC, 0, MODE_RES_F32,
             xn2, nullptr, xnf, nullptr, nullptr, nullptr, nullptr};
  gemm_wmma_kernel<<<dim3(TOK / 32, CC / 32), 32, 0, stream>>>(gp);
  // motion = (cr - ce) @ mproj.T + b -> out_mot f32
  gp = GemmP{mob, wmp, mproj_b, TOK, MD, MD, 0, MODE_OUT_F32,
             out_mot, nullptr, nullptr, nullptr, nullptr, nullptr, nullptr};
  gemm_wmma_kernel<<<dim3(TOK / 32, MD / 32), 32, 0, stream>>>(gp);

  // BN2 + MLP
  bn2_stats_kernel<<<CC, 256, 0, stream>>>(xn2, st2m, st2r);
  bn2_apply_kernel<<<TOK * CC / 256, 256, 0, stream>>>(xn2, st2m, st2r, g2, b2, xbb);

  gp = GemmP{xbb, wf1, fc1_b, TOK, HIDD, CC, 0, MODE_DW_PRELU,
             nullptr, hb, nullptr, dw_w, dw_b, prelu, nullptr};
  gemm_wmma_kernel<<<dim3(TOK / 32, HIDD / 32), 32, 0, stream>>>(gp);

  gp = GemmP{hb, wf2, fc2_b, TOK, CC, HIDD, 0, MODE_FINAL_X,
             out_x, nullptr, nullptr, nullptr, nullptr, nullptr, x};
  gemm_wmma_kernel<<<dim3(TOK / 32, CC / 32), 32, 0, stream>>>(gp);
}
